// Model_27015344292115
// MI455X (gfx1250) — matrix operand order repaired and checked by reading, not compiled, not run on hardware
//
#include <hip/hip_runtime.h>
#include <hip/hip_bf16.h>

// ---------------------------------------------------------------------------
// Shapes: B=256, C=256, S=512, L=256  (C == L)
// Pipeline (per batch b), all 256x256 GEMMs done with v_wmma_f32_16x16x32_f16:
//   x_de = Wc0@Xe + Wc1@Xo + M@Xl + cb        (fused convT+conv)
//   RevIN norm (per-row mean/std)  -> xn (f16)
//   scores = codenorm - 2 * xn@codebook^T ; idx = argmin ; trend = xn - code[idx]
//   h = relu(W1@trend + b1) ; out = (W2@h + b2 + code[idx]) * std + mean
// VQ codebook staging: double-buffered gfx1250 async global->LDS DMA (ASYNCcnt),
// DMA of chunk k+1 overlapped with WMMA of chunk k.
// ---------------------------------------------------------------------------

#define Bn   256
#define Cn   256
#define Sn   512
#define Ln   256
#define KP   40          // padded LDS K-stride (f16): 80B rows -> 16B aligned b128 reads
#define EPSf 1e-5f

typedef _Float16 v8h  __attribute__((ext_vector_type(8)));
typedef _Float16 v16h __attribute__((ext_vector_type(16)));
typedef float    v8f  __attribute__((ext_vector_type(8)));

// ---------------- WMMA helpers -------------------------------------------------

__device__ __forceinline__ v8f wmma16x16x32(v16h a, v16h b, v8f c) {
  // D = A(16x32 f16) * B(32x16 f16) + C(16x16 f32)
  return __builtin_amdgcn_wmma_f32_16x16x32_f16(false, a, false, b, (short)0, c,
                                                false, false);
}

// A-operand (16x32, row-major source, ld = 256):
// lane<16 : row = lane,  K {0..7} and {16..23}; lane>=16: row = lane-16, K {8..15},{24..31}
__device__ __forceinline__ v16h load_a_frag(const _Float16* __restrict__ A,
                                            int row, int kabs) {
  union { v16h v; v8h h[2]; } u;
  const v8h* p0 = (const v8h*)(A + (size_t)row * 256 + kabs);
  const v8h* p1 = (const v8h*)(A + (size_t)row * 256 + kabs + 16);
  u.h[0] = *p0;
  u.h[1] = *p1;
  return u.v;
}

// B-operand from LDS BT[N][KP] (BT[n][k] = B[k][n]):
// lane<16 : col = n0+lane, K {0..7},{16..23} ; lane>=16 : col, K {8..15},{24..31}
__device__ __forceinline__ v16h load_b_frag(const _Float16* __restrict__ BT,
                                            int col, int kb) {
  union { v16h v; v8h h[2]; } u;
  const v8h* p0 = (const v8h*)(BT + col * KP + (kb >> 1));
  const v8h* p1 = (const v8h*)(BT + col * KP + (kb >> 1) + 16);
  u.h[0] = *p0;
  u.h[1] = *p1;
  return u.v;
}

__device__ __forceinline__ void st_pair(_Float16* __restrict__ BT, int n, int kp,
                                        float lo, float hi) {
  union { _Float16 h[2]; unsigned int u; } p;
  p.h[0] = (_Float16)lo;
  p.h[1] = (_Float16)hi;
  *(unsigned int*)(BT + n * KP + 2 * kp) = p.u;
}

// Stage BT[n][k] = src[k][n] from an f16 row-major [K=256][N=256] matrix (K-chunk kc..kc+31).
__device__ __forceinline__ void stage_bt_transpose_f16(const _Float16* __restrict__ src,
                                                       int kc, _Float16* __restrict__ BT) {
  const int t  = threadIdx.x;
  const int kp = t & 15;          // k-pair 0..15
  const int n0 = (t >> 4) * 16;   // 16 columns
  const int r0 = kc + 2 * kp;
  union { uint4 u[2]; _Float16 h[16]; } a, b;
  const uint4* q0 = (const uint4*)(src + (size_t)r0 * 256 + n0);
  const uint4* q1 = (const uint4*)(src + (size_t)(r0 + 1) * 256 + n0);
  a.u[0] = q0[0]; a.u[1] = q0[1];
  b.u[0] = q1[0]; b.u[1] = q1[1];
#pragma unroll
  for (int j = 0; j < 16; ++j) {
    union { _Float16 h[2]; unsigned int u; } p;
    p.h[0] = a.h[j];
    p.h[1] = b.h[j];
    *(unsigned int*)(BT + (n0 + j) * KP + 2 * kp) = p.u;
  }
}

// ---- gfx1250 async global->LDS DMA (no VGPR round-trip), tracked by ASYNCcnt --
__device__ __forceinline__ void async_ld_b128(const void* gaddr, void* ldsaddr) {
  asm volatile("global_load_async_to_lds_b128 %0, %1, off"
               :
               : "v"((unsigned int)(uintptr_t)ldsaddr),
                 "v"((unsigned long long)(uintptr_t)gaddr)
               : "memory");
}
__device__ __forceinline__ void wait_asynccnt0() {
  asm volatile("s_wait_asynccnt 0x0" ::: "memory");
}

// Issue async DMA: BT[n][k] = src[n][kc+k] (direct copy into padded LDS rows).
__device__ __forceinline__ void issue_bt_copy_async(const _Float16* __restrict__ src,
                                                    int kc, _Float16* __restrict__ BT) {
  const int n = threadIdx.x;
  const _Float16* g = src + (size_t)n * 256 + kc;   // 32 f16 = 64B contiguous
  _Float16* l = BT + n * KP;                        // padded row, 16B aligned
#pragma unroll
  for (int q = 0; q < 4; ++q) {
    async_ld_b128(g + q * 8, l + q * 8);
  }
}

// ---------------- Kernel 1: weight prep ---------------------------------------
// M[o,e] = sum_{c,j} Wc[o,c,j] * Wt[c,e,j] ;  cb[o] = bc[o] + sum_{c,j} Wc[o,c,j]*bt[c]
__global__ void __launch_bounds__(256) prep_kernel(
    const float* __restrict__ Wc, const float* __restrict__ Wt,
    const float* __restrict__ bc, const float* __restrict__ bt,
    const float* __restrict__ cbk, const float* __restrict__ W1,
    const float* __restrict__ W2,
    _Float16* __restrict__ Wc0h, _Float16* __restrict__ Wc1h,
    _Float16* __restrict__ Mh, float* __restrict__ cbf,
    _Float16* __restrict__ W1h, _Float16* __restrict__ W2h,
    _Float16* __restrict__ cbkh, float* __restrict__ codenorm) {
  const int o = blockIdx.x;
  const int t = threadIdx.x;
  const int i = o * 256 + t;
  Wc0h[i] = (_Float16)Wc[2 * i + 0];
  Wc1h[i] = (_Float16)Wc[2 * i + 1];
  W1h[i]  = (_Float16)W1[i];
  W2h[i]  = (_Float16)W2[i];
  const float cv = cbk[i];
  cbkh[i] = (_Float16)cv;

  float m = 0.f;
  for (int c = 0; c < 256; ++c) {
    const float w0 = Wc[(o * 256 + c) * 2 + 0];
    const float w1 = Wc[(o * 256 + c) * 2 + 1];
    m += w0 * Wt[(c * 256 + t) * 2 + 0] + w1 * Wt[(c * 256 + t) * 2 + 1];
  }
  Mh[i] = (_Float16)m;

  __shared__ float red[256];
  red[t] = cv * cv;
  __syncthreads();
  for (int s = 128; s > 0; s >>= 1) {
    if (t < s) red[t] += red[t + s];
    __syncthreads();
  }
  if (t == 0) {
    codenorm[o] = red[0];
    float s = bc[o];
    for (int c = 0; c < 256; ++c)
      s += (Wc[(o * 256 + c) * 2 + 0] + Wc[(o * 256 + c) * 2 + 1]) * bt[c];
    cbf[o] = s;
  }
}

// ---------------- Kernel 2: fused convT+conv GEMM ------------------------------
// x_de[b] = Wc0@Xe + Wc1@Xo + M@Xl + cb   ; Xe/Xo deinterleaved from x on the fly.
__global__ void __launch_bounds__(256) gemm_xde_kernel(
    const float* __restrict__ x, const float* __restrict__ xlow,
    const _Float16* __restrict__ Wc0h, const _Float16* __restrict__ Wc1h,
    const _Float16* __restrict__ Mh, const float* __restrict__ cbf,
    float* __restrict__ x_de) {
  __shared__ _Float16 BTe[256 * KP];
  __shared__ _Float16 BTo[256 * KP];
  __shared__ _Float16 BTl[256 * KP];

  const int b    = blockIdx.y;
  const int mblk = blockIdx.x * 64;
  const int lane = threadIdx.x & 31;
  const int w    = threadIdx.x >> 5;
  const int wm   = w >> 2;   // 0..1
  const int wn   = w & 3;    // 0..3

  v8f acc[2][4];
#pragma unroll
  for (int mi = 0; mi < 2; ++mi)
#pragma unroll
    for (int ni = 0; ni < 4; ++ni) acc[mi][ni] = (v8f)0.f;

  for (int kc = 0; kc < 256; kc += 32) {
    // ---- stage Xe/Xo (f32 deinterleave->f16) and Xl into transposed LDS tiles
    {
      const int t  = threadIdx.x;
      const int kp = t & 15;
      const int n0 = (t >> 4) * 16;
      const int r0 = kc + 2 * kp;
      const float4* xe0 = (const float4*)(x + ((size_t)b * 256 + r0) * 512 + 2 * n0);
      const float4* xe1 = (const float4*)(x + ((size_t)b * 256 + r0 + 1) * 512 + 2 * n0);
#pragma unroll
      for (int jj = 0; jj < 8; ++jj) {
        float4 u0 = xe0[jj], u1 = xe1[jj];
        const int na = n0 + 2 * jj;
        st_pair(BTe, na,     kp, u0.x, u1.x);
        st_pair(BTo, na,     kp, u0.y, u1.y);
        st_pair(BTe, na + 1, kp, u0.z, u1.z);
        st_pair(BTo, na + 1, kp, u0.w, u1.w);
      }
      const float4* xl0 = (const float4*)(xlow + ((size_t)b * 256 + r0) * 256 + n0);
      const float4* xl1 = (const float4*)(xlow + ((size_t)b * 256 + r0 + 1) * 256 + n0);
#pragma unroll
      for (int jj = 0; jj < 4; ++jj) {
        float4 u0 = xl0[jj], u1 = xl1[jj];
        st_pair(BTl, n0 + 4 * jj + 0, kp, u0.x, u1.x);
        st_pair(BTl, n0 + 4 * jj + 1, kp, u0.y, u1.y);
        st_pair(BTl, n0 + 4 * jj + 2, kp, u0.z, u1.z);
        st_pair(BTl, n0 + 4 * jj + 3, kp, u0.w, u1.w);
      }
    }
    __syncthreads();

    const int kwin = kc + ((lane >> 4) << 3);
    const int kb   = (lane >> 4) << 4;
    v16h aE[2], aO[2], aL[2];
#pragma unroll
    for (int mi = 0; mi < 2; ++mi) {
      const int row = mblk + (wm * 2 + mi) * 16 + (lane & 15);
      aE[mi] = load_a_frag(Wc0h, row, kwin);
      aO[mi] = load_a_frag(Wc1h, row, kwin);
      aL[mi] = load_a_frag(Mh,   row, kwin);
    }
#pragma unroll
    for (int ni = 0; ni < 4; ++ni) {
      const int col = (wn * 4 + ni) * 16 + (lane & 15);
      v16h bE = load_b_frag(BTe, col, kb);
      v16h bO = load_b_frag(BTo, col, kb);
      v16h bL = load_b_frag(BTl, col, kb);
#pragma unroll
      for (int mi = 0; mi < 2; ++mi) {
        acc[mi][ni] = wmma16x16x32(aE[mi], bE, acc[mi][ni]);
        acc[mi][ni] = wmma16x16x32(aO[mi], bO, acc[mi][ni]);
        acc[mi][ni] = wmma16x16x32(aL[mi], bL, acc[mi][ni]);
      }
    }
    __syncthreads();
  }

  // epilogue: + cb[o], f32 store
#pragma unroll
  for (int mi = 0; mi < 2; ++mi) {
    const int m0 = mblk + (wm * 2 + mi) * 16 + ((lane >> 4) << 3);
#pragma unroll
    for (int ni = 0; ni < 4; ++ni) {
      const int c = (wn * 4 + ni) * 16 + (lane & 15);
#pragma unroll
      for (int i = 0; i < 8; ++i) {
        const int r = m0 + i;
        x_de[(((size_t)b * 256 + r) << 8) + c] = acc[mi][ni][i] + cbf[r];
      }
    }
  }
}

// ---------------- Kernel 3: RevIN normalization --------------------------------
// one wave per (b,c) row: mean/std over L=256, write xn (f16), stats (f32)
__global__ void __launch_bounds__(256) revin_kernel(
    const float* __restrict__ x_de, _Float16* __restrict__ xnh,
    float* __restrict__ meanA, float* __restrict__ stdA) {
  const int lane = threadIdx.x & 31;
  const int w    = threadIdx.x >> 5;
  const int row  = blockIdx.x * 8 + w;      // 0 .. B*C-1
  const size_t base = (size_t)row * 256;

  const float4* p = (const float4*)(x_de + base + lane * 8);
  float4 v0 = p[0], v1 = p[1];
  float s  = v0.x + v0.y + v0.z + v0.w + v1.x + v1.y + v1.z + v1.w;
  float sq = v0.x * v0.x + v0.y * v0.y + v0.z * v0.z + v0.w * v0.w +
             v1.x * v1.x + v1.y * v1.y + v1.z * v1.z + v1.w * v1.w;
#pragma unroll
  for (int off = 16; off > 0; off >>= 1) {
    s  += __shfl_xor(s, off, 32);
    sq += __shfl_xor(sq, off, 32);
  }
  const float mean = s * (1.0f / 256.0f);
  float var = sq * (1.0f / 256.0f) - mean * mean;
  var = var > 0.f ? var : 0.f;
  const float stdv = sqrtf(var + EPSf);
  const float rstd = 1.0f / stdv;

  union { _Float16 h[8]; uint4 u; } o;
  o.h[0] = (_Float16)((v0.x - mean) * rstd);
  o.h[1] = (_Float16)((v0.y - mean) * rstd);
  o.h[2] = (_Float16)((v0.z - mean) * rstd);
  o.h[3] = (_Float16)((v0.w - mean) * rstd);
  o.h[4] = (_Float16)((v1.x - mean) * rstd);
  o.h[5] = (_Float16)((v1.y - mean) * rstd);
  o.h[6] = (_Float16)((v1.z - mean) * rstd);
  o.h[7] = (_Float16)((v1.w - mean) * rstd);
  *(uint4*)(xnh + base + lane * 8) = o.u;
  if (lane == 0) {
    meanA[row] = mean;
    stdA[row]  = stdv;
  }
}

// ---------------- Kernel 4: VQ GEMM + argmin + trend ---------------------------
// score[c,n] = codenorm[n] - 2 * <xn_c, code_n> ; idx=argmin ; trend = xn - code[idx]
// Double-buffered async DMA pipeline: DMA chunk k+1 overlaps WMMA on chunk k.
__global__ void __launch_bounds__(256) vq_kernel(
    const _Float16* __restrict__ xnh, const _Float16* __restrict__ cbkh,
    const float* __restrict__ codenorm, int* __restrict__ idxA,
    _Float16* __restrict__ trendh) {
  __shared__ _Float16 BT0[256 * KP];
  __shared__ _Float16 BT1[256 * KP];
  __shared__ float    sc[64][257];
  __shared__ int      idxsh[64];

  const int b       = blockIdx.y;
  const int rowbase = blockIdx.x * 64;
  const int lane    = threadIdx.x & 31;
  const int w       = threadIdx.x >> 5;
  const int wm      = w >> 2;
  const int wn      = w & 3;
  const _Float16* A = xnh + (size_t)b * 65536;

  v8f acc[2][4];
#pragma unroll
  for (int mi = 0; mi < 2; ++mi)
#pragma unroll
    for (int ni = 0; ni < 4; ++ni) acc[mi][ni] = (v8f)0.f;

  const int kwinrel = (lane >> 4) << 3;
  const int kb      = (lane >> 4) << 4;
  const int r0      = rowbase + (wm * 2 + 0) * 16 + (lane & 15);
  const int r1      = rowbase + (wm * 2 + 1) * 16 + (lane & 15);

  // prologue: issue DMA for chunk 0
  issue_bt_copy_async(cbkh, 0, BT0);

#pragma unroll
  for (int kk = 0; kk < 4; ++kk) {
    // ---- even chunk (2kk) in BT0 ----
    {
      const int kc = (2 * kk) * 32;
      wait_asynccnt0();     // our DMA done
      __syncthreads();      // everyone's DMA done
      issue_bt_copy_async(cbkh, kc + 32, BT1);   // prefetch next, overlaps WMMA
      v16h a0 = load_a_frag(A, r0, kc + kwinrel);
      v16h a1 = load_a_frag(A, r1, kc + kwinrel);
#pragma unroll
      for (int ni = 0; ni < 4; ++ni) {
        const int col = (wn * 4 + ni) * 16 + (lane & 15);
        v16h bf = load_b_frag(BT0, col, kb);
        acc[0][ni] = wmma16x16x32(a0, bf, acc[0][ni]);
        acc[1][ni] = wmma16x16x32(a1, bf, acc[1][ni]);
      }
    }
    // ---- odd chunk (2kk+1) in BT1 ----
    {
      const int kc = (2 * kk + 1) * 32;
      wait_asynccnt0();
      __syncthreads();
      if (kk < 3) issue_bt_copy_async(cbkh, kc + 32, BT0);
      v16h a0 = load_a_frag(A, r0, kc + kwinrel);
      v16h a1 = load_a_frag(A, r1, kc + kwinrel);
#pragma unroll
      for (int ni = 0; ni < 4; ++ni) {
        const int col = (wn * 4 + ni) * 16 + (lane & 15);
        v16h bf = load_b_frag(BT1, col, kb);
        acc[0][ni] = wmma16x16x32(a0, bf, acc[0][ni]);
        acc[1][ni] = wmma16x16x32(a1, bf, acc[1][ni]);
      }
    }
  }
  __syncthreads();

  // scores -> LDS
#pragma unroll
  for (int mi = 0; mi < 2; ++mi) {
    const int l0 = (wm * 2 + mi) * 16 + ((lane >> 4) << 3);
#pragma unroll
    for (int ni = 0; ni < 4; ++ni) {
      const int c = (wn * 4 + ni) * 16 + (lane & 15);
#pragma unroll
      for (int i = 0; i < 8; ++i)
        sc[l0 + i][c] = codenorm[c] - 2.0f * acc[mi][ni][i];
    }
  }
  __syncthreads();

  // per-row argmin (wave w owns rows w*8 .. w*8+7)
  for (int q = 0; q < 8; ++q) {
    const int lr = w * 8 + q;
    float best = 3.4e38f;
    int   bi   = 0;
#pragma unroll
    for (int j = 0; j < 8; ++j) {
      const int n = lane * 8 + j;
      const float v = sc[lr][n];
      if (v < best) { best = v; bi = n; }
    }
#pragma unroll
    for (int off = 16; off > 0; off >>= 1) {
      const float ov = __shfl_xor(best, off, 32);
      const int   oi = __shfl_xor(bi, off, 32);
      if (ov < best || (ov == best && oi < bi)) { best = ov; bi = oi; }
    }
    if (lane == 0) {
      idxsh[lr] = bi;
      idxA[b * 256 + rowbase + lr] = bi;
    }
  }
  __syncthreads();

  // trend = xn - codebook[idx]   (f16 vector math)
  {
    const int t   = threadIdx.x;
    const int lr  = t >> 2;
    const int l0  = (t & 3) * 64;
    const int id  = idxsh[lr];
    const size_t gr = (size_t)(b * 256 + rowbase + lr) * 256;
    const uint4* xp = (const uint4*)(xnh + gr + l0);
    const uint4* cp = (const uint4*)(cbkh + (size_t)id * 256 + l0);
    uint4* tp = (uint4*)(trendh + gr + l0);
#pragma unroll
    for (int q = 0; q < 8; ++q) {
      union { uint4 u; v8h h; } ux, uc, ut;
      ux.u = xp[q];
      uc.u = cp[q];
      ut.h = ux.h - uc.h;
      tp[q] = ut.u;
    }
  }
}

// ---------------- Kernel 5: MLP layer 1 (relu) ---------------------------------
__global__ void __launch_bounds__(256) mlp1_kernel(
    const _Float16* __restrict__ W1h, const _Float16* __restrict__ trendh,
    const float* __restrict__ b1, _Float16* __restrict__ hh) {
  __shared__ _Float16 BT[256 * KP];
  const int b    = blockIdx.y;
  const int mblk = blockIdx.x * 64;
  const int lane = threadIdx.x & 31;
  const int w    = threadIdx.x >> 5;
  const int wm   = w >> 2;
  const int wn   = w & 3;
  const _Float16* Bsrc = trendh + (size_t)b * 65536;

  v8f acc[2][4];
#pragma unroll
  for (int mi = 0; mi < 2; ++mi)
#pragma unroll
    for (int ni = 0; ni < 4; ++ni) acc[mi][ni] = (v8f)0.f;

  for (int kc = 0; kc < 256; kc += 32) {
    stage_bt_transpose_f16(Bsrc, kc, BT);
    __syncthreads();
    const int kwin = kc + ((lane >> 4) << 3);
    const int kb   = (lane >> 4) << 4;
    v16h a0 = load_a_frag(W1h, mblk + (wm * 2 + 0) * 16 + (lane & 15), kwin);
    v16h a1 = load_a_frag(W1h, mblk + (wm * 2 + 1) * 16 + (lane & 15), kwin);
#pragma unroll
    for (int ni = 0; ni < 4; ++ni) {
      const int col = (wn * 4 + ni) * 16 + (lane & 15);
      v16h bf = load_b_frag(BT, col, kb);
      acc[0][ni] = wmma16x16x32(a0, bf, acc[0][ni]);
      acc[1][ni] = wmma16x16x32(a1, bf, acc[1][ni]);
    }
    __syncthreads();
  }

#pragma unroll
  for (int mi = 0; mi < 2; ++mi) {
    const int m0 = mblk + (wm * 2 + mi) * 16 + ((lane >> 4) << 3);
#pragma unroll
    for (int ni = 0; ni < 4; ++ni) {
      const int c = (wn * 4 + ni) * 16 + (lane & 15);
#pragma unroll
      for (int i = 0; i < 8; ++i) {
        const int r = m0 + i;
        float v = acc[mi][ni][i] + b1[r];
        v = v > 0.f ? v : 0.f;
        hh[(((size_t)b * 256 + r) << 8) + c] = (_Float16)v;
      }
    }
  }
}

// ---------------- Kernel 6: MLP layer 2 + quant add + RevIN denorm (fused) -----
__global__ void __launch_bounds__(256) mlp2_final_kernel(
    const _Float16* __restrict__ W2h, const _Float16* __restrict__ hh,
    const float* __restrict__ b2, const float* __restrict__ cbk,
    const int* __restrict__ idxA, const float* __restrict__ stdA,
    const float* __restrict__ meanA, float* __restrict__ out) {
  __shared__ _Float16 BT[256 * KP];
  const int b    = blockIdx.y;
  const int mblk = blockIdx.x * 64;
  const int lane = threadIdx.x & 31;
  const int w    = threadIdx.x >> 5;
  const int wm   = w >> 2;
  const int wn   = w & 3;
  const _Float16* Bsrc = hh + (size_t)b * 65536;

  v8f acc[2][4];
#pragma unroll
  for (int mi = 0; mi < 2; ++mi)
#pragma unroll
    for (int ni = 0; ni < 4; ++ni) acc[mi][ni] = (v8f)0.f;

  for (int kc = 0; kc < 256; kc += 32) {
    stage_bt_transpose_f16(Bsrc, kc, BT);
    __syncthreads();
    const int kwin = kc + ((lane >> 4) << 3);
    const int kb   = (lane >> 4) << 4;
    v16h a0 = load_a_frag(W2h, mblk + (wm * 2 + 0) * 16 + (lane & 15), kwin);
    v16h a1 = load_a_frag(W2h, mblk + (wm * 2 + 1) * 16 + (lane & 15), kwin);
#pragma unroll
    for (int ni = 0; ni < 4; ++ni) {
      const int col = (wn * 4 + ni) * 16 + (lane & 15);
      v16h bf = load_b_frag(BT, col, kb);
      acc[0][ni] = wmma16x16x32(a0, bf, acc[0][ni]);
      acc[1][ni] = wmma16x16x32(a1, bf, acc[1][ni]);
    }
    __syncthreads();
  }

#pragma unroll
  for (int mi = 0; mi < 2; ++mi) {
    const int m0 = mblk + (wm * 2 + mi) * 16 + ((lane >> 4) << 3);
#pragma unroll
    for (int ni = 0; ni < 4; ++ni) {
      const int c = (wn * 4 + ni) * 16 + (lane & 15);
#pragma unroll
      for (int i = 0; i < 8; ++i) {
        const int r  = m0 + i;
        const int gr = b * 256 + r;
        const int id = idxA[gr];
        const float quant  = cbk[(size_t)id * 256 + c];
        const float trend2 = acc[mi][ni][i] + b2[r];
        out[((size_t)gr << 8) + c] = (trend2 + quant) * stdA[gr] + meanA[gr];
      }
    }
  }
}

// ---------------- host-side launch --------------------------------------------

extern "C" void kernel_launch(void* const* d_in, const int* in_sizes, int n_in,
                              void* d_out, int out_size, void* d_ws, size_t ws_size,
                              hipStream_t stream) {
  (void)in_sizes; (void)n_in; (void)out_size; (void)ws_size;
  const float* x   = (const float*)d_in[0];
  const float* xlw = (const float*)d_in[1];
  const float* Wc  = (const float*)d_in[2];
  const float* bc  = (const float*)d_in[3];
  const float* Wt  = (const float*)d_in[4];
  const float* bt  = (const float*)d_in[5];
  const float* cbk = (const float*)d_in[6];
  const float* W1  = (const float*)d_in[7];
  const float* b1  = (const float*)d_in[8];
  const float* W2  = (const float*)d_in[9];
  const float* b2  = (const float*)d_in[10];
  float* out = (float*)d_out;

  // workspace carve-up (256B aligned)
  char* W = (char*)d_ws;
  size_t off = 0;
  auto carve = [&](size_t bytes) -> void* {
    void* p = W + off;
    off = (off + bytes + 255) & ~(size_t)255;
    return p;
  };
  _Float16* Wc0h     = (_Float16*)carve(65536 * 2);
  _Float16* Wc1h     = (_Float16*)carve(65536 * 2);
  _Float16* Mh       = (_Float16*)carve(65536 * 2);
  _Float16* W1h      = (_Float16*)carve(65536 * 2);
  _Float16* W2h      = (_Float16*)carve(65536 * 2);
  _Float16* cbkh     = (_Float16*)carve(65536 * 2);
  float*    cbf      = (float*)carve(256 * 4);
  float*    codenorm = (float*)carve(256 * 4);
  float*    meanA    = (float*)carve(65536 * 4);
  float*    stdA     = (float*)carve(65536 * 4);
  int*      idxA     = (int*)carve(65536 * 4);
  float*    x_de     = (float*)carve((size_t)Bn * 65536 * 4);   // 64 MB
  _Float16* xnh      = (_Float16*)carve((size_t)Bn * 65536 * 2);// 32 MB
  _Float16* trendh   = (_Float16*)carve((size_t)Bn * 65536 * 2);// 32 MB
  _Float16* hh       = (_Float16*)carve((size_t)Bn * 65536 * 2);// 32 MB

  prep_kernel<<<256, 256, 0, stream>>>(Wc, Wt, bc, bt, cbk, W1, W2,
                                       Wc0h, Wc1h, Mh, cbf, W1h, W2h, cbkh,
                                       codenorm);
  gemm_xde_kernel<<<dim3(4, Bn), 256, 0, stream>>>(x, xlw, Wc0h, Wc1h, Mh, cbf,
                                                   x_de);
  revin_kernel<<<(Bn * Cn) / 8, 256, 0, stream>>>(x_de, xnh, meanA, stdA);
  vq_kernel<<<dim3(4, Bn), 256, 0, stream>>>(xnh, cbkh, codenorm, idxA, trendh);
  mlp1_kernel<<<dim3(4, Bn), 256, 0, stream>>>(W1h, trendh, b1, hh);
  mlp2_final_kernel<<<dim3(4, Bn), 256, 0, stream>>>(W2h, hh, b2, cbk, idxA,
                                                     stdA, meanA, out);
}
